// MBartGQAttention_68839735820314
// MI455X (gfx1250) — matrix operand order repaired and checked by reading, not compiled, not run on hardware
//
#include <hip/hip_runtime.h>
#include <hip/hip_bf16.h>

// ---------------------------------------------------------------------------
// MBart GQA attention for MI455X (gfx1250).
// All matmuls via v_wmma_f32_16x16x32_bf16 (bf16 in, f32 accumulate).
// Roofline: ~56 GFLOP vs ~45MB DRAM traffic -> compute-bound -> WMMA path.
// GEMMs register-blocked 32x64 per wave (2x4 accumulators, no spills).
// Softmax row-max uses v_permlane16_b32 (pure VALU butterfly, no LDS);
// softmax row-sum is a WMMA against a ones matrix (reuses the P A-frag).
// ---------------------------------------------------------------------------

typedef __bf16 bf16_t;
typedef __attribute__((ext_vector_type(8)))  bf16_t v8bf;
typedef __attribute__((ext_vector_type(16))) bf16_t v16bf;
typedef __attribute__((ext_vector_type(8)))  float  v8f;

#define WB    2
#define WT    2048
#define WE    1024
#define WH    16
#define WKVH  4
#define WD    64
#define WM    (WB * WT)            // 4096 GEMM rows
#define WSCALE 0.125f              // D^-0.5, D=64
#define LOG2E 1.44269504088896340736f

static __device__ __forceinline__ bf16_t f2bf(float f) {
  unsigned u = __builtin_bit_cast(unsigned, f);
  u += 0x7FFFu + ((u >> 16) & 1u);             // round-to-nearest-even
  unsigned short h = (unsigned short)(u >> 16);
  return __builtin_bit_cast(bf16_t, h);
}

// permlane16 selector: lane (base+i) reads lane ((base+i)^m) within its row
static constexpr unsigned permsel(int m, int base) {
  unsigned v = 0;
  for (int i = 0; i < 8; ++i)
    v |= (unsigned)(((base + i) ^ m) & 0xF) << (4 * i);
  return v;
}

// xor-butterfly lane swap within 16-lane rows, pure VALU (v_permlane16_b32)
template <int MASK>
static __device__ __forceinline__ float xor16(float x) {
  constexpr unsigned s0 = permsel(MASK, 0);
  constexpr unsigned s1 = permsel(MASK, 8);
  unsigned xi = __builtin_bit_cast(unsigned, x);
  unsigned yi = __builtin_amdgcn_permlane16(xi, xi, s0, s1, false, false);
  return __builtin_bit_cast(float, yi);
}

// A-fragment, 16x32 bf16, source row-major with leading dim `ld` (halves).
// Lane L: row = L&15, elements 0..7 = K[koff..koff+7], 8..15 = K[koff+16..+23],
// koff = (L>=16)?8:0  (ISA 7.12.2 16-bit A layout). Two 16B loads.
static __device__ __forceinline__ v16bf load_fragA(const bf16_t* base, int ld,
                                                   int k0, int lane) {
  const int m    = lane & 15;
  const int koff = (lane >> 4) << 3;
  const bf16_t* p = base + (size_t)m * ld + k0 + koff;
  v8bf lo = *(const v8bf*)(p);
  v8bf hi = *(const v8bf*)(p + 16);
  v16bf r;
#pragma unroll
  for (int i = 0; i < 8; ++i) { r[i] = lo[i]; r[i + 8] = hi[i]; }
  return r;
}

// B-fragment, 32x16 bf16, source stored "transposed": column n's K-values are
// contiguous (row n of `base`, leading dim `ld` halves).
// Lane L: col = L&15, elements 0..7 = K[basek..basek+7], 8..15 = K[basek+16..+23],
// basek = (L>=16)?8:0.
static __device__ __forceinline__ v16bf load_fragB(const bf16_t* base, int ld,
                                                   int k0, int lane) {
  const int n     = lane & 15;
  const int basek = (lane >> 4) << 3;
  const bf16_t* p = base + (size_t)n * ld + k0 + basek;
  v8bf lo = *(const v8bf*)(p);
  v8bf hi = *(const v8bf*)(p + 16);
  v16bf r;
#pragma unroll
  for (int i = 0; i < 8; ++i) { r[i] = lo[i]; r[i + 8] = hi[i]; }
  return r;
}

static __device__ __forceinline__ v8f wmma_bf16(v16bf a, v16bf b, v8f c) {
  return __builtin_amdgcn_wmma_f32_16x16x32_bf16(
      /*neg_a=*/false, a, /*neg_b=*/false, b,
      /*c_mod=*/(short)0, c, /*reuse_a=*/false, /*reuse_b=*/false);
}

// ---------------------------------------------------------------------------
// fp32 -> bf16 convert (grid-stride)
// ---------------------------------------------------------------------------
__global__ void cvt_f32_bf16(const float* __restrict__ in,
                             bf16_t* __restrict__ out, int nv) {
  int i = blockIdx.x * blockDim.x + threadIdx.x;
  int stride = gridDim.x * blockDim.x;
  for (; i < nv; i += stride) out[i] = f2bf(in[i]);
}

// ---------------------------------------------------------------------------
// WMMA GEMM: C[WM,N] = A[WM,K] @ W[N,K]^T + bias.   (all shapes compile-time)
// 32x64 tile per wave: 2x4 v8f accumulators, 2 A-frags + 4 B-frags per
// 32-wide K step. launch_bounds(256,1) gives the RA a 1-wave/SIMD budget.
//   MODE 0: out bf16 at [B, Hn, T, D]  (val = (acc+bias)*scale)   -> Q / K
//   MODE 1: out bf16 at [B, Hn, D, T]  (transposed V for PV GEMM) -> V
//   MODE 2: out f32  at [WM, N]        (final output projection)
// ---------------------------------------------------------------------------
template <int N, int K, int MODE>
__global__ void __launch_bounds__(256, 1)
wmma_gemm_bf16(const bf16_t* __restrict__ A,
               const bf16_t* __restrict__ W,
               const float* __restrict__ bias,
               float scale, int Hn,
               bf16_t* __restrict__ obf,
               float* __restrict__ of32) {
  const int lane = threadIdx.x & 31;
  const int wv   = threadIdx.x >> 5;
  constexpr int NTILES = N >> 6;                // 64-wide N tiles
  const int wt = blockIdx.x * 8 + wv;
  const int mt = wt / NTILES, nt = wt % NTILES; // power-of-two -> shifts
  const int m0 = mt << 5, n0 = nt << 6;
  const bf16_t* Ab = A + (size_t)m0 * K;
  const bf16_t* Wb = W + (size_t)n0 * K;

  v8f acc[2][4];
#pragma unroll
  for (int i = 0; i < 2; ++i)
#pragma unroll
    for (int j = 0; j < 4; ++j) acc[i][j] = (v8f){};

  for (int k0 = 0; k0 < K; k0 += 32) {
    // prefetch next K chunk of this wave's operand rows (global_prefetch_b8)
    __builtin_prefetch(Ab + (size_t)(lane & 15) * K + k0 + 128, 0, 1);
    __builtin_prefetch(Wb + (size_t)(lane & 15) * K + k0 + 128, 0, 1);
    v16bf a[2], b[4];
#pragma unroll
    for (int i = 0; i < 2; ++i)
      a[i] = load_fragA(Ab + (size_t)(i * 16) * K, K, k0, lane);
#pragma unroll
    for (int j = 0; j < 4; ++j)
      b[j] = load_fragB(Wb + (size_t)(j * 16) * K, K, k0, lane);
#pragma unroll
    for (int i = 0; i < 2; ++i)
#pragma unroll
      for (int j = 0; j < 4; ++j)
        acc[i][j] = wmma_bf16(a[i], b[j], acc[i][j]);
  }

  const int half = lane >> 4;
#pragma unroll
  for (int j = 0; j < 4; ++j) {
    const int n  = n0 + j * 16 + (lane & 15);
    const float bv = bias[n];
    if (MODE == 2) {
#pragma unroll
      for (int i = 0; i < 2; ++i)
#pragma unroll
        for (int r = 0; r < 8; ++r) {
          const int m = m0 + i * 16 + r + (half << 3);
          of32[(size_t)m * N + n] = acc[i][j][r] + bv;
        }
    } else {
      const int hh = n >> 6, d = n & (WD - 1);   // WD = 64
#pragma unroll
      for (int i = 0; i < 2; ++i)
#pragma unroll
        for (int r = 0; r < 8; ++r) {
          const int m  = m0 + i * 16 + r + (half << 3);
          const int bb = m >> 11, t = m & (WT - 1);  // WT = 2048
          const float val = (acc[i][j][r] + bv) * scale;
          size_t addr;
          if (MODE == 0)
            addr = (((size_t)bb * Hn + hh) * WT + t) * WD + d;
          else
            addr = (((size_t)bb * Hn + hh) * WD + d) * (size_t)WT + t;
          obf[addr] = f2bf(val);
        }
    }
  }
}

// ---------------------------------------------------------------------------
// Flash attention: one wave = one (b, h, 16-row Q block). Streams S in chunks
// of 32 with online softmax. q: [B,H,T,D], k: [B,KVH,T,D], vt: [B,KVH,D,T].
// Output aout: [B,T,E] bf16 with head order h = kvh*G + g.
// ---------------------------------------------------------------------------
__global__ void __launch_bounds__(256, 1)
wmma_flash_attn(const bf16_t* __restrict__ q,
                const bf16_t* __restrict__ kk,
                const bf16_t* __restrict__ vt,
                bf16_t* __restrict__ aout) {
  __shared__ __align__(16) bf16_t Pl[8][16 * 32];   // per-wave P staging, 1KB

  const int lane = threadIdx.x & 31;
  const int wv   = threadIdx.x >> 5;
  const int wid  = blockIdx.x * 8 + wv;
  const int qblk = wid & 127;           // T/16 = 128 q blocks
  const int h    = (wid >> 7) & 15;
  const int b    = wid >> 11;
  const int kvh  = h >> 2;              // G = 4
  const int t0   = qblk << 4;

  const bf16_t* qh = q  + (((size_t)(b * WH + h)) * WT + t0) * WD;
  const bf16_t* kh = kk + ((size_t)(b * WKVH + kvh)) * WT * WD;
  const bf16_t* vh = vt + ((size_t)(b * WKVH + kvh)) * WD * WT;

  // Q fragments for D=64 (two K-steps of 32), kept resident
  const v16bf qa0 = load_fragA(qh, WD, 0, lane);
  const v16bf qa1 = load_fragA(qh, WD, 32, lane);

  // all-ones B fragment: row-sum of P = P @ Ones (replicated across lanes)
  v16bf ones;
#pragma unroll
  for (int i = 0; i < 16; ++i) ones[i] = f2bf(1.0f);

  v8f accO[4];
#pragma unroll
  for (int dt = 0; dt < 4; ++dt) accO[dt] = (v8f){};
  float Mr[8], Lr[8], alph[8];
#pragma unroll
  for (int r = 0; r < 8; ++r) { Mr[r] = -__builtin_inff(); Lr[r] = 0.0f; }

  bf16_t* Pw = &Pl[wv][0];
  const int n    = lane & 15;
  const int half = lane >> 4;

  for (int s0 = 0; s0 < WT; s0 += 32) {
    // ---- scores: two 16x16 tiles (cols s0..s0+15 and s0+16..s0+31) ----
    v8f sc0 = {}, sc1 = {};
    {
      const bf16_t* kb0 = kh + (size_t)s0 * WD;
      sc0 = wmma_bf16(qa0, load_fragB(kb0, WD, 0, lane), sc0);
      sc0 = wmma_bf16(qa1, load_fragB(kb0, WD, 32, lane), sc0);
      const bf16_t* kb1 = kh + (size_t)(s0 + 16) * WD;
      sc1 = wmma_bf16(qa0, load_fragB(kb1, WD, 0, lane), sc1);
      sc1 = wmma_bf16(qa1, load_fragB(kb1, WD, 32, lane), sc1);
    }

    // ---- online softmax: row-max via permlane16 butterfly (pure VALU) ----
#pragma unroll
    for (int r = 0; r < 8; ++r) {
      float rm = fmaxf(sc0[r], sc1[r]);
      rm = fmaxf(rm, xor16<1>(rm));
      rm = fmaxf(rm, xor16<2>(rm));
      rm = fmaxf(rm, xor16<4>(rm));
      rm = fmaxf(rm, xor16<8>(rm));
      const float mn = fmaxf(Mr[r], rm);
      alph[r] = exp2f((Mr[r] - mn) * LOG2E);
      Mr[r]   = mn;
      const float p0 = exp2f((sc0[r] - mn) * LOG2E);
      const float p1 = exp2f((sc1[r] - mn) * LOG2E);
#pragma unroll
      for (int dt = 0; dt < 4; ++dt) accO[dt][r] *= alph[r];
      // stage P tile (C-layout -> row-major LDS) for A-fragment reload
      const int mrow = r + (half << 3);
      Pw[mrow * 32 + n]      = f2bf(p0);
      Pw[mrow * 32 + n + 16] = f2bf(p1);
    }

    // in-wave LDS RAW: DS ops are in-order per wave; fence the compiler too
    asm volatile("s_wait_dscnt 0x0" ::: "memory");

    // ---- PV: P(16x32) @ V(32x64); row-sum: P @ Ones (one extra wmma) ----
    const v16bf pa = load_fragA(Pw, 32, 0, lane);
    const v8f rsum = wmma_bf16(pa, ones, (v8f){});
#pragma unroll
    for (int r = 0; r < 8; ++r) Lr[r] = Lr[r] * alph[r] + rsum[r];
#pragma unroll
    for (int dt = 0; dt < 4; ++dt) {
      const v16bf vb = load_fragB(vh + (size_t)(dt * 16) * WT, WT, s0, lane);
      accO[dt] = wmma_bf16(pa, vb, accO[dt]);
    }
  }

  // ---- normalize and store [B,T,E] bf16 ----
#pragma unroll
  for (int r = 0; r < 8; ++r) {
    const float inv = 1.0f / Lr[r];
    const int t = t0 + r + (half << 3);
    const size_t rowbase = ((size_t)b * WT + t) * WE + (size_t)h * WD;
#pragma unroll
    for (int dt = 0; dt < 4; ++dt)
      aout[rowbase + dt * 16 + n] = f2bf(accO[dt][r] * inv);
  }
}

// ---------------------------------------------------------------------------
// Launch
// ---------------------------------------------------------------------------
extern "C" void kernel_launch(void* const* d_in, const int* in_sizes, int n_in,
                              void* d_out, int out_size, void* d_ws,
                              size_t ws_size, hipStream_t stream) {
  (void)in_sizes; (void)n_in; (void)out_size; (void)ws_size;
  const float* x  = (const float*)d_in[0];
  const float* Wq = (const float*)d_in[1];
  const float* bq = (const float*)d_in[2];
  const float* Wk = (const float*)d_in[3];
  const float* bk = (const float*)d_in[4];
  const float* Wv = (const float*)d_in[5];
  const float* bv = (const float*)d_in[6];
  const float* Wo = (const float*)d_in[7];
  const float* bo = (const float*)d_in[8];
  float* out = (float*)d_out;

  char* w = (char*)d_ws;
  auto alloc = [&](size_t bytes) -> char* {
    char* p = w; w += (bytes + 255) & ~(size_t)255; return p;
  };
  bf16_t* xb  = (bf16_t*)alloc((size_t)WB * WT * WE * 2);          // 8 MB
  bf16_t* Wqb = (bf16_t*)alloc((size_t)WE * WE * 2);               // 2 MB
  bf16_t* Wkb = (bf16_t*)alloc((size_t)WKVH * WD * WE * 2);        // 512 KB
  bf16_t* Wvb = (bf16_t*)alloc((size_t)WKVH * WD * WE * 2);        // 512 KB
  bf16_t* Wob = (bf16_t*)alloc((size_t)WE * WE * 2);               // 2 MB
  bf16_t* qb  = (bf16_t*)alloc((size_t)WB * WH * WT * WD * 2);     // 8 MB
  bf16_t* kb  = (bf16_t*)alloc((size_t)WB * WKVH * WT * WD * 2);   // 2 MB
  bf16_t* vtb = (bf16_t*)alloc((size_t)WB * WKVH * WD * WT * 2);   // 2 MB
  bf16_t* ab  = (bf16_t*)alloc((size_t)WB * WT * WE * 2);          // 8 MB

  const dim3 blk(256);
  cvt_f32_bf16<<<1024, blk, 0, stream>>>(x,  xb,  WB * WT * WE);
  cvt_f32_bf16<<<512,  blk, 0, stream>>>(Wq, Wqb, WE * WE);
  cvt_f32_bf16<<<256,  blk, 0, stream>>>(Wk, Wkb, WKVH * WD * WE);
  cvt_f32_bf16<<<256,  blk, 0, stream>>>(Wv, Wvb, WKVH * WD * WE);
  cvt_f32_bf16<<<512,  blk, 0, stream>>>(Wo, Wob, WE * WE);

  const int wavesQ = (WM / 32) * (WE / 64);            // 2048 waves
  const int wavesK = (WM / 32) * ((WKVH * WD) / 64);   // 512 waves

  // Q = (x @ Wq^T + bq) * SCALE -> [B,H,T,D]
  wmma_gemm_bf16<WE, WE, 0><<<wavesQ / 8, blk, 0, stream>>>(
      xb, Wqb, bq, WSCALE, WH, qb, nullptr);
  // K = x @ Wk^T + bk -> [B,KVH,T,D]
  wmma_gemm_bf16<WKVH * WD, WE, 0><<<wavesK / 8, blk, 0, stream>>>(
      xb, Wkb, bk, 1.0f, WKVH, kb, nullptr);
  // V = x @ Wv^T + bv -> transposed [B,KVH,D,T]
  wmma_gemm_bf16<WKVH * WD, WE, 1><<<wavesK / 8, blk, 0, stream>>>(
      xb, Wvb, bv, 1.0f, WKVH, vtb, nullptr);
  // attention: 2*16*128 = 4096 waves
  wmma_flash_attn<<<(WB * WH * (WT / 16)) / 8, blk, 0, stream>>>(qb, kb, vtb, ab);
  // out = attn @ Wo^T + bo (f32)
  wmma_gemm_bf16<WE, WE, 2><<<wavesQ / 8, blk, 0, stream>>>(
      ab, Wob, bo, 1.0f, WH, nullptr, out);
}
